// GRIT_attention_67886253081128
// MI455X (gfx1250) — matrix operand order repaired and checked by reading, not compiled, not run on hardware
//
#include <hip/hip_runtime.h>
#include <hip/hip_bf16.h>
#include <stdint.h>

// GRIT attention, fused for MI455X (gfx1250, wave32, WMMA, async-to-LDS).
// B=4, N=512, C=128, E=32, E2=64, A=8.
//
// Memory-bound: edge(134MB)+embedding(134MB) in, edge_out(134MB) out
// => ~17us floor at 23.3 TB/s. ~8 GFLOP of GEMM is negligible vs WMMA
// throughput, so everything is fused into one pass over the edge tensor;
// QK (268MB) and edge_new (134MB) never touch HBM.

typedef _Float16 v16h __attribute__((ext_vector_type(16)));
typedef float    v8f  __attribute__((ext_vector_type(8)));
typedef float    v4f  __attribute__((ext_vector_type(4)));

#define GB  4
#define GN  512
#define GC  128
#define GE  32
#define GE2 64
#define GA  8

// ---- async global->LDS support (gfx1250 GLOBAL_LOAD_ASYNC_TO_LDS_B128) ----
#if defined(__has_builtin)
#if __has_builtin(__builtin_amdgcn_global_load_async_to_lds_b128) && \
    __has_builtin(__builtin_amdgcn_s_wait_asynccnt)
#define USE_ASYNC_LDS 1
#endif
#endif
#ifndef USE_ASYNC_LDS
#define USE_ASYNC_LDS 0
#endif

#if USE_ASYNC_LDS
// builtin signature: (global int4* src, local int4* dst, imm offset, imm cpol)
typedef int b128_t __attribute__((vector_size(16)));
typedef __attribute__((address_space(3))) b128_t lds_b128_t;
typedef __attribute__((address_space(1))) b128_t glb_b128_t;
__device__ __forceinline__ lds_b128_t* to_lds(void* p) {
  // generic LDS address: addr[31:0] is the LDS byte offset (ISA 10.2 aperture rules)
  return (lds_b128_t*)(uint32_t)(uintptr_t)p;
}
__device__ __forceinline__ glb_b128_t* to_glb(const void* p) {
  return (glb_b128_t*)(uintptr_t)p;
}
#endif

__device__ __forceinline__ int lane_id() { return (int)(threadIdx.x & 31); }

// ---- WMMA fragment builders (layouts per CDNA5 ISA 7.12.2) ----

// 16-bit A matrix 16x32 (MxK) from a row-major f16 tile in LDS.
// lane l: m = l&15, hi = l>>4; VGPR g pair: k = (g&3)*2 + (g>>2)*16 + hi*8 + {0,1}
__device__ __forceinline__ v16h frag_A_lds(const _Float16* tile, int stride) {
  const int l = lane_id();
  const int m = l & 15;
  const int hi = l >> 4;
  v16h a;
#pragma unroll
  for (int g = 0; g < 8; ++g) {
    const int kb = ((g & 3) << 1) + ((g >> 2) << 4) + (hi << 3);
    a[2 * g]     = tile[m * stride + kb];
    a[2 * g + 1] = tile[m * stride + kb + 1];
  }
  return a;
}

// Same A fragment, but source tile is f32 in LDS (convert during build).
__device__ __forceinline__ v16h frag_A_ldsf(const float* tile, int stride) {
  const int l = lane_id();
  const int m = l & 15;
  const int hi = l >> 4;
  v16h a;
#pragma unroll
  for (int g = 0; g < 8; ++g) {
    const int kb = ((g & 3) << 1) + ((g >> 2) << 4) + (hi << 3);
    a[2 * g]     = (_Float16)tile[m * stride + kb];
    a[2 * g + 1] = (_Float16)tile[m * stride + kb + 1];
  }
  return a;
}

// B matrix 32x16 (KxN) built from row-major weights W[n][k] (B = W^T slice).
// lane l: n = n0 + (l&15), hi = l>>4; VGPR g: k = (g&3)*2 + (g>>2)*16 + hi*8 + {0,1}
__device__ __forceinline__ v16h frag_B_lds(const _Float16* W, int ld, int n0) {
  const int l = lane_id();
  const int n = n0 + (l & 15);
  const int hi = l >> 4;
  v16h b;
#pragma unroll
  for (int g = 0; g < 8; ++g) {
    const int k = ((g & 3) << 1) + ((g >> 2) << 4) + (hi << 3);
    b[2 * g]     = W[n * ld + k];
    b[2 * g + 1] = W[n * ld + k + 1];
  }
  return b;
}

__device__ __forceinline__ v16h frag_B_glb(const float* W, int ld, int n0, int k0) {
  const int l = lane_id();
  const int n = n0 + (l & 15);
  const int hi = l >> 4;
  v16h b;
#pragma unroll
  for (int g = 0; g < 8; ++g) {
    const int k = k0 + ((g & 3) << 1) + ((g >> 2) << 4) + (hi << 3);
    b[2 * g]     = (_Float16)W[n * ld + k];
    b[2 * g + 1] = (_Float16)W[n * ld + k + 1];
  }
  return b;
}

__device__ __forceinline__ v8f wmma_f16(v16h a, v16h b, v8f c) {
  // 8 args: (neg_a, A, neg_b, B, c_mod, C, reuse_a, reuse_b)
  return __builtin_amdgcn_wmma_f32_16x16x32_f16(false, a, false, b, (short)0, c, false, false);
}

__device__ __forceinline__ float rho_f(float x) {
  return copysignf(sqrtf(fabsf(x)), x);
}
__device__ __forceinline__ float gelu_f(float x) {
  return 0.5f * x * (1.0f + erff(x * 0.70710678118654752f));
}

// ============================================================
// Kernel 1: Q/K/V projection.  node (2048x128) x {Wq^T,Wk^T (128x512), Wv^T (128x128)}
// One 16-row M-tile per block; 8 waves sweep the 72 16-col N-tiles.
// ============================================================
__global__ __launch_bounds__(256) void grit_qkv(
    const float* __restrict__ node,
    const float* __restrict__ Wq, const float* __restrict__ bq,
    const float* __restrict__ Wk, const float* __restrict__ bk,
    const float* __restrict__ Wv, const float* __restrict__ bv,
    float* __restrict__ Qf, float* __restrict__ Kf, float* __restrict__ Vf) {
  __shared__ _Float16 sA[16 * 128];
  const int tid = threadIdx.x;
  const int row0 = blockIdx.x * 16;

  for (int idx = tid; idx < 16 * 128; idx += 256)
    sA[idx] = (_Float16)node[(size_t)(row0 + (idx >> 7)) * GC + (idx & 127)];
  __syncthreads();

  const int w = tid >> 5;
  const int l = tid & 31;
  const int n_l = l & 15;
  const int hi = l >> 4;

  for (int nt = w; nt < 72; nt += 8) {
    const float* W; const float* bias; float* out; int n0; int ldo;
    if (nt < 32)      { W = Wq; bias = bq; out = Qf; n0 = nt * 16;        ldo = GE2 * GA; }
    else if (nt < 64) { W = Wk; bias = bk; out = Kf; n0 = (nt - 32) * 16; ldo = GE2 * GA; }
    else              { W = Wv; bias = bv; out = Vf; n0 = (nt - 64) * 16; ldo = GC; }
    v8f acc = {};
#pragma unroll
    for (int ks = 0; ks < 4; ++ks) {
      v16h a = frag_A_lds(sA + ks * 32, 128);
      v16h b = frag_B_glb(W, GC, n0, ks * 32);
      acc = wmma_f16(a, b, acc);
    }
    const int n = n0 + n_l;
#pragma unroll
    for (int r = 0; r < 8; ++r) {
      const int m = r + (hi << 3);
      out[(size_t)(row0 + m) * ldo + n] = acc[r] + bias[n];
    }
  }
}

// ============================================================
// Kernel 2: fully fused edge/attention pass.
// One workgroup (128 threads = 4 waves) per (b, i) row:
//   - edge/embedding 2KB tiles pulled in with GLOBAL_LOAD_ASYNC_TO_LDS_B128
//     (ASYNCcnt), waves free-run over their tiles
//   - QK[b,i,j,:] computed on the fly (A=8 VALU reduction, K from L2)
//   - edge@Wew.T, edge@Web.T via WMMA f16 per 16-j tile
//   - edge_new kept in LDS as f16 (32KB of the 320KB/WGP)
//   - row softmax, agg, alpha@V, node-out GEMM chain
//   - edge_out = edge_new @ Weo.T via WMMA, streamed out non-temporally
// ============================================================
__global__ __launch_bounds__(128) void grit_edge(
    const float* __restrict__ edge, const float* __restrict__ emb,
    const float* __restrict__ Qf, const float* __restrict__ Kf,
    const float* __restrict__ Vf,
    const float* __restrict__ Wew, const float* __restrict__ Web,
    const float* __restrict__ Wen, const float* __restrict__ ben,
    const float* __restrict__ Wa,  const float* __restrict__ Wno,
    const float* __restrict__ bno, const float* __restrict__ Weo,
    float* __restrict__ outNode, float* __restrict__ outEdge) {
  __shared__ _Float16 sEN[GN * GE];          // 32 KB edge_new row (f16)
  __shared__ float    sQ[GE2 * GA];          // Q[i,a,e] (2 KB)
  __shared__ float    sScore[GN];            // scores -> alpha (2 KB)
  __shared__ _Float16 sWew[GE * GE], sWeb[GE * GE], sWeo[GE * GE];
  __shared__ float    sWaS[GE];
  __shared__ float    sStageF[4][16 * GE];   // per-wave edge tiles (f32, 8 KB)
  __shared__ float    sEmbF[4][16 * GE];     // per-wave embedding tiles (f32, 8 KB)
  __shared__ float    sRed[128];
  __shared__ float    sMS[2];
  __shared__ float    sAggP[4][GE];
  __shared__ float    sAgg[GE];
  __shared__ float    sPre[GC];

  const int tid = threadIdx.x;
  const int w   = tid >> 5;
  const int l   = tid & 31;
  const int n   = l & 15;
  const int hi  = l >> 4;

  const int bi = blockIdx.x;                 // b*512 + i
  const int b  = bi >> 9;
  const size_t qoff    = (size_t)bi * (GE2 * GA);        // elem offset of Q row
  const size_t kbase   = (size_t)b * GN;                 // row base into Kf/Vf
  const size_t edgeRow = (size_t)bi * GN * GE;           // elem offset of edge[b,i,0,0]

  for (int t = tid; t < GE2 * GA; t += 128) sQ[t] = Qf[qoff + t];
  for (int t = tid; t < GE * GE; t += 128) {
    sWew[t] = (_Float16)Wew[t];
    sWeb[t] = (_Float16)Web[t];
    sWeo[t] = (_Float16)Weo[t];
  }
  if (tid < GE) sWaS[tid] = Wa[tid];
  __syncthreads();

  // per-wave resident B fragments (weights reused for all 32 j-tiles)
  const v16h bW0 = frag_B_lds(sWew, GE, 0),  bW1 = frag_B_lds(sWew, GE, 16);
  const v16h bB0 = frag_B_lds(sWeb, GE, 0),  bB1 = frag_B_lds(sWeb, GE, 16);
  const v16h bO0 = frag_B_lds(sWeo, GE, 0),  bO1 = frag_B_lds(sWeo, GE, 16);

  // ---- main pass over 32 j-tiles; each wave owns tiles it*4+w, free-running ----
  for (int it = 0; it < 8; ++it) {
    const int jt0 = (it * 4 + w) * 16;

    // stage this wave's contiguous 2KB edge + embedding tiles into LDS
#if USE_ASYNC_LDS
    {
      const char* gE = (const char*)(edge + edgeRow + (size_t)jt0 * GE);
      const char* gM = (const char*)(emb  + edgeRow + (size_t)jt0 * GE);
      char* sE = (char*)&sStageF[w][0];
      char* sM = (char*)&sEmbF[w][0];
#pragma unroll
      for (int q = 0; q < 4; ++q) {
        const int off = q * 512 + l * 16;
        __builtin_amdgcn_global_load_async_to_lds_b128(to_glb(gE + off), to_lds(sE + off), 0, 0);
        __builtin_amdgcn_global_load_async_to_lds_b128(to_glb(gM + off), to_lds(sM + off), 0, 0);
      }
      __builtin_amdgcn_s_wait_asynccnt(0);
    }
#else
    {
      const int jrow = l >> 1;
      const int e0   = (l & 1) * 16;
      const float* srcE = edge + edgeRow + (size_t)(jt0 + jrow) * GE + e0;
      const float* srcM = emb  + edgeRow + (size_t)(jt0 + jrow) * GE + e0;
      if (it < 7) __builtin_prefetch(srcE + 64 * GE, 0, 0);
      float* dE = &sStageF[w][jrow * GE + e0];
      float* dM = &sEmbF[w][jrow * GE + e0];
#pragma unroll
      for (int q = 0; q < 4; ++q) {
        const v4f ve = __builtin_nontemporal_load((const v4f*)(srcE + 4 * q));
        const v4f vm = __builtin_nontemporal_load((const v4f*)(srcM + 4 * q));
#pragma unroll
        for (int c = 0; c < 4; ++c) { dE[4 * q + c] = ve[c]; dM[4 * q + c] = vm[c]; }
      }
    }
#endif

    const v16h a = frag_A_ldsf(sStageF[w], GE);  // K=32 covered by one WMMA
    v8f cw0 = {}, cw1 = {}, cb0 = {}, cb1 = {};
    cw0 = wmma_f16(a, bW0, cw0);   // (edge @ Wew.T), e = 0..15
    cw1 = wmma_f16(a, bW1, cw1);   //                 e = 16..31
    cb0 = wmma_f16(a, bB0, cb0);   // (edge @ Web.T)
    cb1 = wmma_f16(a, bB1, cb1);

#pragma unroll
    for (int r = 0; r < 8; ++r) {
      const int m = r + (hi << 3);
      const int j = jt0 + m;
      const float* Krow = Kf + (kbase + j) * (GE2 * GA);
      // QK[b,i,j,e] = sum_a Q[i,a,e] * K[j,a,e];  qk_w: e<32, qk_b: e>=32
      float qkw0 = 0.f, qkb0 = 0.f, qkw1 = 0.f, qkb1 = 0.f;
#pragma unroll
      for (int aa = 0; aa < GA; ++aa) {
        const int base = aa * GE2;
        qkw0 += sQ[base + n]      * Krow[base + n];
        qkw1 += sQ[base + 16 + n] * Krow[base + 16 + n];
        qkb0 += sQ[base + 32 + n] * Krow[base + 32 + n];
        qkb1 += sQ[base + 48 + n] * Krow[base + 48 + n];
      }
      const float e0v = sEmbF[w][m * GE + n];
      const float e1v = sEmbF[w][m * GE + 16 + n];
      const float v0 = rho_f(qkw0 * cw0[r]) + cb0[r] + qkb0 + e0v;
      const float v1 = rho_f(qkw1 * cw1[r]) + cb1[r] + qkb1 + e1v;
      const float en0 = gelu_f(v0);
      const float en1 = gelu_f(v1);
      sEN[j * GE + n]      = (_Float16)en0;
      sEN[j * GE + 16 + n] = (_Float16)en1;
      // score contribution for softmax: dot(edge_new[j,:], Wa)
      float contrib = en0 * sWaS[n] + en1 * sWaS[16 + n];
#pragma unroll
      for (int off = 1; off < 16; off <<= 1)
        contrib += __shfl_xor(contrib, off, 32);
      if (n == 0) sScore[j] = contrib;
    }
  }
  __syncthreads();

  // ---- softmax over j (row of 512) ----
  {
    float lm = -3.0e38f;
    for (int j = tid; j < GN; j += 128) lm = fmaxf(lm, sScore[j]);
    sRed[tid] = lm;
    __syncthreads();
    if (tid == 0) {
      float m = -3.0e38f;
      for (int k = 0; k < 128; ++k) m = fmaxf(m, sRed[k]);
      sMS[0] = m;
    }
    __syncthreads();
    const float mx = sMS[0];
    float ls = 0.f;
    for (int j = tid; j < GN; j += 128) {
      const float e = expf(sScore[j] - mx);
      sScore[j] = e;
      ls += e;
    }
    sRed[tid] = ls;
    __syncthreads();
    if (tid == 0) {
      float s = 0.f;
      for (int k = 0; k < 128; ++k) s += sRed[k];
      sMS[1] = 1.0f / s;
    }
    __syncthreads();
    const float inv = sMS[1];
    for (int j = tid; j < GN; j += 128) sScore[j] *= inv;   // now alpha
    __syncthreads();
  }

  // ---- agg[e] = sum_j alpha[j] * edge_new[j,e] ----
  {
    const int e = tid & 31;
    const int q = tid >> 5;
    float s = 0.f;
    for (int j = q * 128; j < (q + 1) * 128; ++j)
      s += sScore[j] * (float)sEN[j * GE + e];
    sAggP[q][e] = s;
  }
  __syncthreads();
  if (tid < GE) sAgg[tid] = sAggP[0][tid] + sAggP[1][tid] + sAggP[2][tid] + sAggP[3][tid];
  __syncthreads();

  // ---- node path: alpha@V + agg@Wen.T + ben, then @Wno.T + bno ----
  {
    const int c = tid;                                // 128 threads = C
    float accv = 0.f;
    for (int j = 0; j < GN; ++j)
      accv += sScore[j] * Vf[(kbase + j) * GC + c];   // V resident in L2
    float t = 0.f;
#pragma unroll
    for (int e = 0; e < GE; ++e) t += sAgg[e] * Wen[c * GE + e];
    sPre[c] = accv + t + ben[c];
  }
  __syncthreads();
  {
    const int c2 = tid;
    float o = bno[c2];
    for (int c = 0; c < GC; ++c) o += Wno[c2 * GC + c] * sPre[c];
    outNode[(size_t)bi * GC + c2] = o;
  }

  // ---- edge_out = edge_new @ Weo.T (WMMA), streamed out non-temporally ----
  for (int it = 0; it < 8; ++it) {
    const int jt0 = (it * 4 + w) * 16;
    const v16h ae = frag_A_lds(sEN + jt0 * GE, GE);
    v8f co0 = {}, co1 = {};
    co0 = wmma_f16(ae, bO0, co0);
    co1 = wmma_f16(ae, bO1, co1);
#pragma unroll
    for (int r = 0; r < 8; ++r) {
      const int j = jt0 + r + (hi << 3);
      float* dst = outEdge + edgeRow + (size_t)j * GE;
      __builtin_nontemporal_store(co0[r], dst + n);
      __builtin_nontemporal_store(co1[r], dst + 16 + n);
    }
  }
}

extern "C" void kernel_launch(void* const* d_in, const int* in_sizes, int n_in,
                              void* d_out, int out_size, void* d_ws, size_t ws_size,
                              hipStream_t stream) {
  const float* node      = (const float*)d_in[0];
  const float* edge      = (const float*)d_in[1];
  const float* embedding = (const float*)d_in[2];
  const float* Wq  = (const float*)d_in[3];
  const float* bq  = (const float*)d_in[4];
  const float* Wk  = (const float*)d_in[5];
  const float* bk  = (const float*)d_in[6];
  const float* Wv  = (const float*)d_in[7];
  const float* bv  = (const float*)d_in[8];
  const float* Wew = (const float*)d_in[9];
  const float* Web = (const float*)d_in[10];
  const float* Wen = (const float*)d_in[11];
  const float* ben = (const float*)d_in[12];
  const float* Wa  = (const float*)d_in[13];
  const float* Wno = (const float*)d_in[14];
  const float* bno = (const float*)d_in[15];
  const float* Weo = (const float*)d_in[16];

  // workspace: Q (4MB) | K (4MB) | V (1MB)  -> L2 resident
  float* Qf = (float*)d_ws;
  float* Kf = Qf + (size_t)GB * GN * GE2 * GA;
  float* Vf = Kf + (size_t)GB * GN * GE2 * GA;

  float* outNode = (float*)d_out;
  float* outEdge = outNode + (size_t)GB * GN * GC;

  grit_qkv<<<(GB * GN) / 16, 256, 0, stream>>>(node, Wq, bq, Wk, bk, Wv, bv, Qf, Kf, Vf);
  grit_edge<<<GB * GN, 128, 0, stream>>>(edge, embedding, Qf, Kf, Vf,
                                         Wew, Web, Wen, ben, Wa, Wno, bno, Weo,
                                         outNode, outEdge);
}
